// MultiHeadAttention_26336739459727
// MI455X (gfx1250) — matrix operand order repaired and checked by reading, not compiled, not run on hardware
//
#include <hip/hip_runtime.h>

// ---------------------------------------------------------------------------
// MI455X (gfx1250) multi-head attention: bf16 WMMA + TDM, double-buffered.
// B=4, N=2048, D_MODEL=1024, H=16, DH=64.
// ---------------------------------------------------------------------------

typedef __attribute__((ext_vector_type(16))) __bf16 v16bf;
typedef __attribute__((ext_vector_type(8)))  float  v8f;

union Frag {            // one WMMA 16x16x32 bf16 operand (8 VGPRs)
    v16bf v;
    uint4 q[2];
    unsigned short s[16];
};

__device__ __forceinline__ unsigned short f2bf(float f) {
    union { float f; unsigned u; } v; v.f = f;
    unsigned r = v.u + 0x7FFFu + ((v.u >> 16) & 1u);   // round-to-nearest-even
    return (unsigned short)(r >> 16);
}
__device__ __forceinline__ unsigned pack2(float a, float b) {
    return (unsigned)f2bf(a) | ((unsigned)f2bf(b) << 16);
}
__device__ __forceinline__ v8f zero8() {
    v8f z;
#pragma unroll
    for (int i = 0; i < 8; ++i) z[i] = 0.0f;
    return z;
}
__device__ __forceinline__ v8f wmma_bf16(const Frag& a, const Frag& b, v8f c) {
    return __builtin_amdgcn_wmma_f32_16x16x32_bf16(false, a.v, false, b.v,
                                                   (short)0, c, false, false);
}

constexpr int BATCH = 4;
constexpr int SEQ   = 2048;
constexpr int DM    = 1024;
constexpr int NH    = 16;
constexpr int DH    = 64;
constexpr int MROWS = BATCH * SEQ;   // 8192

// ---------------------------------------------------------------------------
// Tensor Data Mover: 2D bf16 tile copy global -> LDS (D# per ISA ch.8).
// ---------------------------------------------------------------------------
#if __has_builtin(__builtin_amdgcn_tensor_load_to_lds)
#define HAVE_TDM 1
typedef unsigned int v4u __attribute__((ext_vector_type(4)));
typedef int          v8i __attribute__((ext_vector_type(8)));
typedef int          v4i __attribute__((ext_vector_type(4)));

__device__ __forceinline__ void tdm_load_tile_bf16(unsigned lds_off,
                                                   const unsigned short* gptr,
                                                   int tile_rows, int tile_cols,
                                                   int tensor_rows, int tensor_cols,
                                                   int row_stride) {
    unsigned long long ga = (unsigned long long)(size_t)gptr;
    v4u g0;
    g0[0] = 1u;                                           // count=1, user D#
    g0[1] = lds_off;                                      // LDS byte address
    g0[2] = (unsigned)(ga & 0xFFFFFFFFull);               // global_addr[31:0]
    g0[3] = (unsigned)((ga >> 32) & 0x01FFFFFFull) | (2u << 30);  // addr[56:32] | type=2
    v8i g1;
    g1[0] = 1 << 16;                                      // data_size=1 (2 bytes)
    g1[1] = (tensor_cols & 0xFFFF) << 16;                 // tensor_dim0[15:0]
    g1[2] = ((tensor_cols >> 16) & 0xFFFF) | ((tensor_rows & 0xFFFF) << 16);
    g1[3] = ((tensor_rows >> 16) & 0xFFFF) | ((tile_cols & 0xFFFF) << 16);
    g1[4] = (tile_rows & 0xFFFF);                         // tile_dim1 (tile_dim2=0)
    g1[5] = row_stride;                                   // tensor_dim0_stride[31:0]
    g1[6] = 0;
    g1[7] = 0;
    v4i z4 = {0, 0, 0, 0};
#if defined(__clang_major__) && __clang_major__ >= 23
    v8i z8 = {0, 0, 0, 0, 0, 0, 0, 0};
    __builtin_amdgcn_tensor_load_to_lds(g0, g1, z4, z4, z8, 0);
#else
    __builtin_amdgcn_tensor_load_to_lds(g0, g1, z4, z4, 0);
#endif
}
#endif

// ---------------------------------------------------------------------------
// GEMM: C[M x 1024] = A(fp32, M x 1024) @ W(fp32, 1024 x 1024) + bias.
// 128x128 block tile, K-step 32, 8 waves (2x4 WMMA tiles per wave),
// double-buffered LDS staging: next tile's loads overlap current WMMAs.
// SCATTER: write bf16 to head-major [B][H][N][DH]; else fp32 row-major.
// ---------------------------------------------------------------------------
template <bool SCATTER>
__global__ __launch_bounds__(256)
void gemm128(const float* __restrict__ A, const float* __restrict__ W,
             const float* __restrict__ bias, void* __restrict__ outp) {
    constexpr int K = DM, NOUT = DM;
    __shared__ __align__(16) unsigned short lds_a[2][128][32];   // A tiles bf16
    __shared__ __align__(16) unsigned short lds_wt[2][128][32];  // W tiles bf16, [n][k]

    const int tid = threadIdx.x;
    const int n0 = blockIdx.x * 128, m0 = blockIdx.y * 128;
    const int w  = tid >> 5, l = tid & 31, h = l >> 4, ln = l & 15;
    const int wr = w & 3, wc = w >> 2;     // wave -> 32-row x 64-col sub-block

    auto stage = [&](int bufi, int kk) {
        // ---- A tile (128x32 fp32 -> bf16) ----
#pragma unroll
        for (int i = 0; i < 4; ++i) {
            int c = tid + 256 * i;            // 1024 chunks of 4 floats
            int row = c >> 3, c4 = (c & 7) * 4;
            const float* ap = A + (size_t)(m0 + row) * K + kk + c4;
            const float4 f = *(const float4*)ap;
            if (kk + 32 < K) __builtin_prefetch(ap + 32, 0, 3);   // tile after next
            uint2 u; u.x = pack2(f.x, f.y); u.y = pack2(f.z, f.w);
            *(uint2*)&lds_a[bufi][row][c4] = u;
        }
        // ---- W tile transposed (32x128 fp32 -> bf16 [n][k]) ----
#pragma unroll
        for (int i = 0; i < 4; ++i) {
            int c = tid + 256 * i;
            int kr = c >> 5, c4 = (c & 31) * 4;
            const float4 f = *(const float4*)(W + (size_t)(kk + kr) * NOUT + n0 + c4);
            lds_wt[bufi][c4 + 0][kr] = f2bf(f.x);
            lds_wt[bufi][c4 + 1][kr] = f2bf(f.y);
            lds_wt[bufi][c4 + 2][kr] = f2bf(f.z);
            lds_wt[bufi][c4 + 3][kr] = f2bf(f.w);
        }
    };

    v8f acc[2][4];
#pragma unroll
    for (int mi = 0; mi < 2; ++mi)
#pragma unroll
        for (int nc = 0; nc < 4; ++nc) acc[mi][nc] = zero8();

    stage(0, 0);                                   // prologue
    for (int kk = 0; kk < K; kk += 32) {
        const int cur = (kk >> 5) & 1;
        __syncthreads();                           // cur tile visible; nxt buf free
        if (kk + 32 < K) stage(cur ^ 1, kk + 32);  // overlaps with WMMAs below

        Frag af[2], bf[4];
#pragma unroll
        for (int mi = 0; mi < 2; ++mi) {
            int row = wr * 32 + mi * 16 + ln;         // M = lane%16
            af[mi].q[0] = *(const uint4*)&lds_a[cur][row][8 * h];        // K 8h..8h+7
            af[mi].q[1] = *(const uint4*)&lds_a[cur][row][8 * h + 16];   // K 8h+16..
        }
#pragma unroll
        for (int nc = 0; nc < 4; ++nc) {
            int col = wc * 64 + nc * 16 + ln;         // N = lane%16
            bf[nc].q[0] = *(const uint4*)&lds_wt[cur][col][8 * h];       // K 8h..8h+7
            bf[nc].q[1] = *(const uint4*)&lds_wt[cur][col][8 * h + 16];  // K 8h+16..
        }
#pragma unroll
        for (int mi = 0; mi < 2; ++mi)
#pragma unroll
            for (int nc = 0; nc < 4; ++nc)
                acc[mi][nc] = wmma_bf16(af[mi], bf[nc], acc[mi][nc]);
    }

    // ---- epilogue: bias + store ----
#pragma unroll
    for (int mi = 0; mi < 2; ++mi)
#pragma unroll
        for (int nc = 0; nc < 4; ++nc) {
            int col = n0 + wc * 64 + nc * 16 + ln;
            float bv = bias[col];
#pragma unroll
            for (int r = 0; r < 8; ++r) {
                int row = m0 + wr * 32 + mi * 16 + r + 8 * h;
                float val = acc[mi][nc][r] + bv;
                if (SCATTER) {
                    int head = col >> 6, dh = col & 63;
                    int bb = row >> 11, n = row & 2047;
                    ((unsigned short*)outp)[(((size_t)bb * NH + head) * SEQ + n) * DH + dh] = f2bf(val);
                } else {
                    ((float*)outp)[(size_t)row * DM + col] = val;
                }
            }
        }
}

// ---------------------------------------------------------------------------
// Flash attention per (b, head): 64 queries/block (4 waves x 16 rows),
// streams 64-key tiles, double-buffered. K tile via Tensor Data Mover
// (one TDM slice per wave, TENSORcnt), V tile staged transposed so the
// B-operand fragments are contiguous ds_load_b128 pairs. TDM + V staging
// for tile i+1 overlap the WMMA/softmax work on tile i.
// ---------------------------------------------------------------------------
__global__ __launch_bounds__(128)
void attn_kernel(const unsigned short* __restrict__ Qb,
                 const unsigned short* __restrict__ Kb,
                 const unsigned short* __restrict__ Vb,
                 const float* __restrict__ mask,
                 float* __restrict__ Obuf) {
    __shared__ __align__(16) unsigned short k_lds[2][64][64];    // [key][d]
    __shared__ __align__(16) unsigned short vt_lds[2][64][64];   // [d][key]
    __shared__ __align__(16) unsigned short p_lds[4][16][64];    // per-wave P scratch

    const int tid = threadIdx.x;
    const int w = tid >> 5, l = tid & 31, h = l >> 4, ln = l & 15;
    const int bh = blockIdx.y;                 // b*16 + head
    const int b = bh >> 4, head = bh & 15;
    const int q0 = blockIdx.x * 64;
    const float* maskb = mask + (size_t)b * SEQ * SEQ;

    auto stage = [&](int bufi, int ktn) {
#ifdef HAVE_TDM
        // K tile: wave w DMAs its 16 rows via the Tensor Data Mover.
        tdm_load_tile_bf16((unsigned)(size_t)&k_lds[bufi][w * 16][0],
                           Kb + ((size_t)bh * SEQ + ktn + w * 16) * DH,
                           /*tile_rows=*/16, /*tile_cols=*/DH,
                           /*tensor_rows=*/SEQ, /*tensor_cols=*/DH,
                           /*row_stride=*/DH);
#endif
        int row = tid >> 1, hf = tid & 1;
#ifndef HAVE_TDM
        const uint4* kg = (const uint4*)(Kb + ((size_t)bh * SEQ + ktn + row) * DH + hf * 32);
#pragma unroll
        for (int j = 0; j < 4; ++j)
            *(uint4*)&k_lds[bufi][row][hf * 32 + j * 8] = kg[j];
#endif
        const uint4* vg = (const uint4*)(Vb + ((size_t)bh * SEQ + ktn + row) * DH + hf * 32);
#pragma unroll
        for (int j = 0; j < 4; ++j) {
            union { uint4 q; unsigned short s[8]; } u;
            u.q = vg[j];
#pragma unroll
            for (int e = 0; e < 8; ++e)
                vt_lds[bufi][hf * 32 + j * 8 + e][row] = u.s[e];
        }
        // prefetch the mask strip for tile ktn (computed next iteration)
        __builtin_prefetch(maskb + (size_t)(q0 + row) * SEQ + ktn + hf * 32, 0, 3);
    };

    // Q fragments for this wave's 16 rows (A-layout), loaded once.
    Frag qf[2];
    {
        const unsigned short* qr = Qb + ((size_t)bh * SEQ + (q0 + w * 16 + ln)) * DH;
#pragma unroll
        for (int ks = 0; ks < 2; ++ks) {
            qf[ks].q[0] = *(const uint4*)(qr + ks * 32 + h * 8);
            qf[ks].q[1] = *(const uint4*)(qr + ks * 32 + h * 8 + 16);
        }
    }

    v8f oa[4];
#pragma unroll
    for (int ds = 0; ds < 4; ++ds) oa[ds] = zero8();
    float mrow[8], lrow[8];
#pragma unroll
    for (int r = 0; r < 8; ++r) { mrow[r] = -1e30f; lrow[r] = 0.0f; }

    stage(0, 0);                               // prologue: tile 0 in flight

    for (int kt = 0; kt < SEQ; kt += 64) {
        const int cur = (kt >> 6) & 1;
#ifdef HAVE_TDM
        __builtin_amdgcn_s_wait_tensorcnt(0);  // own TDM slice for cur landed
#endif
        __syncthreads();                       // all waves' cur K + V visible
        if (kt + 64 < SEQ) stage(cur ^ 1, kt + 64);  // overlap with compute below

        // ---- S = Q K^T (16 rows x 64 keys per wave) ----
        v8f st[4];
#pragma unroll
        for (int nc = 0; nc < 4; ++nc) st[nc] = zero8();
#pragma unroll
        for (int ks = 0; ks < 2; ++ks)
#pragma unroll
            for (int nc = 0; nc < 4; ++nc) {
                Frag bfK;
                int krow = nc * 16 + ln;
                bfK.q[0] = *(const uint4*)&k_lds[cur][krow][ks * 32 + h * 8];
                bfK.q[1] = *(const uint4*)&k_lds[cur][krow][ks * 32 + h * 8 + 16];
                st[nc] = wmma_bf16(qf[ks], bfK, st[nc]);
            }

        // ---- scale + multiplicative mask, row max ----
        float mx[8];
#pragma unroll
        for (int r = 0; r < 8; ++r) mx[r] = -1e30f;
#pragma unroll
        for (int nc = 0; nc < 4; ++nc) {
            int kcol = kt + nc * 16 + ln;
#pragma unroll
            for (int r = 0; r < 8; ++r) {
                int qrow = q0 + w * 16 + r + 8 * h;
                float mval = maskb[(size_t)qrow * SEQ + kcol];
                float sv = st[nc][r] * 0.125f * mval;
                st[nc][r] = sv;
                mx[r] = fmaxf(mx[r], sv);
            }
        }
#pragma unroll
        for (int r = 0; r < 8; ++r) {
            mx[r] = fmaxf(mx[r], __shfl_xor(mx[r], 1, 32));
            mx[r] = fmaxf(mx[r], __shfl_xor(mx[r], 2, 32));
            mx[r] = fmaxf(mx[r], __shfl_xor(mx[r], 4, 32));
            mx[r] = fmaxf(mx[r], __shfl_xor(mx[r], 8, 32));
        }

        // ---- online softmax update ----
        float sc[8], rs[8];
#pragma unroll
        for (int r = 0; r < 8; ++r) {
            float mn = fmaxf(mrow[r], mx[r]);
            sc[r] = __expf(mrow[r] - mn);
            mrow[r] = mn;
            rs[r] = 0.0f;
        }
#pragma unroll
        for (int nc = 0; nc < 4; ++nc)
#pragma unroll
            for (int r = 0; r < 8; ++r) {
                float p = __expf(st[nc][r] - mrow[r]);
                st[nc][r] = p;
                rs[r] += p;
            }
#pragma unroll
        for (int r = 0; r < 8; ++r) {
            rs[r] += __shfl_xor(rs[r], 1, 32);
            rs[r] += __shfl_xor(rs[r], 2, 32);
            rs[r] += __shfl_xor(rs[r], 4, 32);
            rs[r] += __shfl_xor(rs[r], 8, 32);
            lrow[r] = lrow[r] * sc[r] + rs[r];
        }
#pragma unroll
        for (int ds = 0; ds < 4; ++ds)
#pragma unroll
            for (int r = 0; r < 8; ++r) oa[ds][r] *= sc[r];

        // ---- P: C-layout -> LDS bf16 -> A-layout fragments ----
#pragma unroll
        for (int nc = 0; nc < 4; ++nc)
#pragma unroll
            for (int r = 0; r < 8; ++r)
                p_lds[w][r + 8 * h][nc * 16 + ln] = f2bf(st[nc][r]);

        Frag pa[2];
#pragma unroll
        for (int ks = 0; ks < 2; ++ks) {
            pa[ks].q[0] = *(const uint4*)&p_lds[w][ln][ks * 32 + h * 8];
            pa[ks].q[1] = *(const uint4*)&p_lds[w][ln][ks * 32 + h * 8 + 16];
        }

        // ---- O += P V ----
#pragma unroll
        for (int ds = 0; ds < 4; ++ds)
#pragma unroll
            for (int ks = 0; ks < 2; ++ks) {
                Frag vbf;
                int dcol = ds * 16 + ln;
                vbf.q[0] = *(const uint4*)&vt_lds[cur][dcol][ks * 32 + h * 8];
                vbf.q[1] = *(const uint4*)&vt_lds[cur][dcol][ks * 32 + h * 8 + 16];
                oa[ds] = wmma_bf16(pa[ks], vbf, oa[ds]);
            }
    }

    // ---- normalize, write O (fp32, token-major [B][N][DM]) ----
#pragma unroll
    for (int r = 0; r < 8; ++r) {
        float inv = 1.0f / lrow[r];
        int qrow = q0 + w * 16 + r + 8 * h;
        float* orow = Obuf + ((size_t)b * SEQ + qrow) * DM + head * DH;
#pragma unroll
        for (int ds = 0; ds < 4; ++ds)
            orow[ds * 16 + ln] = oa[ds][r] * inv;
    }
}

// ---------------------------------------------------------------------------
extern "C" void kernel_launch(void* const* d_in, const int* in_sizes, int n_in,
                              void* d_out, int out_size, void* d_ws, size_t ws_size,
                              hipStream_t stream) {
    const float* x    = (const float*)d_in[0];
    const float* mask = (const float*)d_in[1];
    const float* Wq   = (const float*)d_in[2];
    const float* bq   = (const float*)d_in[3];
    const float* Wk   = (const float*)d_in[4];
    const float* bk   = (const float*)d_in[5];
    const float* Wv   = (const float*)d_in[6];
    const float* bv   = (const float*)d_in[7];
    const float* Wo   = (const float*)d_in[8];
    const float* bo   = (const float*)d_in[9];

    // Workspace: Qb/Kb/Vb bf16 head-major (16 MB each) + O fp32 (32 MB) = 80 MB.
    unsigned short* Qb = (unsigned short*)d_ws;
    unsigned short* Kb = Qb + (size_t)MROWS * DM;
    unsigned short* Vb = Kb + (size_t)MROWS * DM;
    float*          Ob = (float*)(Vb + (size_t)MROWS * DM);

    dim3 gg(DM / 128, MROWS / 128);   // (8, 64)
    gemm128<true><<<gg, 256, 0, stream>>>(x, Wq, bq, Qb);
    gemm128<true><<<gg, 256, 0, stream>>>(x, Wk, bk, Kb);
    gemm128<true><<<gg, 256, 0, stream>>>(x, Wv, bv, Vb);

    attn_kernel<<<dim3(SEQ / 64, BATCH * NH), 128, 0, stream>>>(Qb, Kb, Vb, mask, Ob);

    gemm128<false><<<gg, 256, 0, stream>>>(Ob, Wo, bo, d_out);
}
